// GraphSAGE_40398462386319
// MI455X (gfx1250) — compile-verified, matrix operand order checked by reading, not run on hardware
//
#include <hip/hip_runtime.h>
#include <hip/hip_bf16.h>

// ---------------------------------------------------------------------------
// GraphSAGE 3-layer (mean aggr) for gfx1250 / MI455X.
//  h_{l+1} = relu( [mean_agg(h_l) || h_l[:Nd]] @ [W_l;W_r] + b )
// Aggregation: counting-sort CSR (int atomics only) + pull-mode register
//              accumulation -> bf16 A build (no f32 scatter atomics).
// GEMM: v_wmma_f32_16x16x32_bf16, K=512 fused (lin_l + lin_r in one pass),
//       software-pipelined operand loads.
// ---------------------------------------------------------------------------

#define NL1 50000
#define NL2 25000
#define NL3 12500
#define E0  800000
#define E1  400000
#define E2  200000
#define HID 256

typedef __attribute__((ext_vector_type(16))) __bf16 v16bf;
typedef __attribute__((ext_vector_type(8)))  float  v8f;

// ---- helpers --------------------------------------------------------------

__device__ __forceinline__ unsigned short f2bf(float f) {
    unsigned u = __float_as_uint(f);
    unsigned r = u + 0x7FFFu + ((u >> 16) & 1u);   // round-to-nearest-even
    return (unsigned short)(r >> 16);
}

// Load one 16-bit WMMA operand fragment for this lane (ISA 7.12.2 layout):
// lanes 0-15 hold K = {0..7, 16..23}, lanes 16-31 hold K = {8..15, 24..31}.
// p already includes row/col*ldk + khalf; reads p[0..7] and p[16..23].
__device__ __forceinline__ v16bf load_frag(const unsigned short* p) {
    union { v16bf v; uint4 u[2]; } r;
    r.u[0] = *(const uint4*)(p);
    r.u[1] = *(const uint4*)(p + 16);
    return r.v;
}

// ---- CSR build ------------------------------------------------------------

__global__ void zero_i32(int* __restrict__ p, int n) {
    int i = blockIdx.x * blockDim.x + threadIdx.x;
    if (i < n) p[i] = 0;
}

__global__ void sage_count(const int* __restrict__ ei, int E, int* __restrict__ cnt) {
    int t = blockIdx.x * blockDim.x + threadIdx.x;
    if (t < E) atomicAdd(&cnt[ei[E + t]], 1);
}

// per-block exclusive scan (256 elems) + block sums
__global__ void scan_block(const int* __restrict__ cnt, int n,
                           int* __restrict__ offs, int* __restrict__ bsums) {
    __shared__ int buf[256];
    int tid = threadIdx.x;
    int i = blockIdx.x * 256 + tid;
    int v = (i < n) ? cnt[i] : 0;
    buf[tid] = v;
    __syncthreads();
#pragma unroll
    for (int d = 1; d < 256; d <<= 1) {
        int t = (tid >= d) ? buf[tid - d] : 0;
        __syncthreads();
        buf[tid] += t;
        __syncthreads();
    }
    if (i < n) offs[i] = buf[tid] - v;            // block-local exclusive
    if (tid == 255) bsums[blockIdx.x] = buf[255];
}

// single block: exclusive scan of up to 1024 block sums (in place)
__global__ void scan_tops(int* __restrict__ bsums, int nb) {
    __shared__ int buf[1024];
    int tid = threadIdx.x;
    int v = (tid < nb) ? bsums[tid] : 0;
    buf[tid] = v;
    __syncthreads();
#pragma unroll
    for (int d = 1; d < 1024; d <<= 1) {
        int t = (tid >= d) ? buf[tid - d] : 0;
        __syncthreads();
        buf[tid] += t;
        __syncthreads();
    }
    if (tid < nb) bsums[tid] = buf[tid] - v;      // exclusive
}

// add block bases; also initialize cursor = offs
__global__ void scan_add(int* __restrict__ offs, int n,
                         const int* __restrict__ bsums, int* __restrict__ cursor) {
    int i = blockIdx.x * 256 + threadIdx.x;
    if (i < n) {
        int v = offs[i] + bsums[i >> 8];
        offs[i]   = v;
        cursor[i] = v;
    }
}

// bucket edges by dst: src_sorted[pos] holds the src node id
__global__ void sage_reorder(const int* __restrict__ ei, int E,
                             int* __restrict__ cursor, int* __restrict__ src_sorted) {
    int t = blockIdx.x * blockDim.x + threadIdx.x;
    if (t >= E) return;
    int dst = ei[E + t];
    int pos = atomicAdd(&cursor[dst], 1);
    src_sorted[pos] = ei[t];
}

// ---- aggregate + bf16 A build ---------------------------------------------
// 64 threads (2 waves) per dst row; float4 accumulation in registers.
// A[row][0:256] = mean, A[row][256:512] = h[row]. Wave-uniform edge loop.
__global__ __launch_bounds__(256)
void sage_aggregate(const float* __restrict__ h, const int* __restrict__ src_sorted,
                    const int* __restrict__ offs, int Nd, int E,
                    unsigned short* __restrict__ A) {
    long gid = (long)blockIdx.x * blockDim.x + threadIdx.x;
    long row = gid >> 6;
    if (row >= Nd) return;
    int q = (int)(gid & 63);
    int start = offs[row];
    int end   = (row == (long)Nd - 1) ? E : offs[row + 1];

    float ax = 0.f, ay = 0.f, az = 0.f, aw = 0.f;
    for (int e = start; e < end; ++e) {
        int src = src_sorted[e];
        float4 v = ((const float4*)(h + (size_t)src * HID))[q];
        ax += v.x; ay += v.y; az += v.z; aw += v.w;
    }
    float inv = 1.0f / fmaxf((float)(end - start), 1.0f);
    float4 hh = ((const float4*)(h + (size_t)row * HID))[q];
    ushort4 a0 = { f2bf(ax * inv), f2bf(ay * inv), f2bf(az * inv), f2bf(aw * inv) };
    ushort4 a1 = { f2bf(hh.x), f2bf(hh.y), f2bf(hh.z), f2bf(hh.w) };
    ((ushort4*)(A + (size_t)row * 512))[q]        = a0;
    ((ushort4*)(A + (size_t)row * 512 + 256))[q]  = a1;
}

// ---- weights: pack [W_l;W_r] (512x256 f32, k-major) -> bf16 n-major -------

__global__ void sage_prepw(const float* __restrict__ Wl, const float* __restrict__ Wr,
                           unsigned short* __restrict__ Wt) {
    int gid = blockIdx.x * blockDim.x + threadIdx.x;
    if (gid >= 256 * 512) return;
    int n = gid >> 9;
    int k = gid & 511;
    float v = (k < 256) ? Wl[(size_t)k * 256 + n] : Wr[(size_t)(k - 256) * 256 + n];
    Wt[(size_t)n * 512 + k] = f2bf(v);
}

// ---- WMMA GEMM ------------------------------------------------------------
// out[M,256] = A[M,512](bf16) @ Wt^T + bias, optional ReLU.
// One wave -> 16 rows x 64 cols (4 accumulators); software-pipelined K loop.
__global__ __launch_bounds__(256)
void sage_gemm(const unsigned short* __restrict__ A,
               const unsigned short* __restrict__ Wt,
               const float* __restrict__ bias,
               float* __restrict__ out,
               int M, int relu) {
    const int lane = threadIdx.x & 31;
    const int wid  = threadIdx.x >> 5;
    const int gw   = blockIdx.x * 8 + wid;
    const int mtiles = (M + 15) >> 4;
    const int mt = gw >> 2;      // m-tile index
    const int ng = gw & 3;       // n-group (64 cols each)
    if (mt >= mtiles) return;    // wave-uniform exit: EXEC stays all-ones

    const int nl   = lane & 15;
    const int kh   = (lane >> 4) * 8;               // K-half per ISA layout
    const int mrow = mt * 16 + nl;
    const int arow = (mrow < M) ? mrow : (M - 1);   // clamp pad rows (store masked)

    const unsigned short* Ap = A + (size_t)arow * 512 + kh;
    const unsigned short* Wn[4];
#pragma unroll
    for (int j = 0; j < 4; ++j)
        Wn[j] = Wt + (size_t)(ng * 64 + j * 16 + nl) * 512 + kh;

    v8f acc[4] = {};

    // prologue: fragments for kk = 0
    v16bf a = load_frag(Ap);
    v16bf b[4];
#pragma unroll
    for (int j = 0; j < 4; ++j) b[j] = load_frag(Wn[j]);

#pragma unroll
    for (int kk = 32; kk <= 512; kk += 32) {
        v16bf an;
        v16bf bn[4];
        if (kk < 512) {                    // prefetch next K-tile
            an = load_frag(Ap + kk);
#pragma unroll
            for (int j = 0; j < 4; ++j) bn[j] = load_frag(Wn[j] + kk);
        }
#pragma unroll
        for (int j = 0; j < 4; ++j)
            acc[j] = __builtin_amdgcn_wmma_f32_16x16x32_bf16(
                false, a, false, b[j], (short)0, acc[j], false, false);
        if (kk < 512) {
            a = an;
#pragma unroll
            for (int j = 0; j < 4; ++j) b[j] = bn[j];
        }
    }

    // C layout: VGPR r <-> M = r + 8*(lane>=16), N = lane&15.
    const int moff = (lane >> 4) * 8;
#pragma unroll
    for (int j = 0; j < 4; ++j) {
        const int n = ng * 64 + j * 16 + nl;
        const float bn = bias[n];
#pragma unroll
        for (int r = 0; r < 8; ++r) {
            int m = mt * 16 + moff + r;
            if (m < M) {
                float v = acc[j][r] + bn;
                if (relu) v = fmaxf(v, 0.0f);
                out[(size_t)m * HID + n] = v;
            }
        }
    }
}

// ---- host-side launch ------------------------------------------------------

static inline long cdivl(long a, long b) { return (a + b - 1) / b; }

extern "C" void kernel_launch(void* const* d_in, const int* in_sizes, int n_in,
                              void* d_out, int out_size, void* d_ws, size_t ws_size,
                              hipStream_t stream) {
    (void)in_sizes; (void)n_in; (void)out_size; (void)ws_size;

    const float* x     = (const float*)d_in[0];
    const int*   ei[3] = { (const int*)d_in[1], (const int*)d_in[2], (const int*)d_in[3] };
    const float* Wl[3] = { (const float*)d_in[4], (const float*)d_in[7], (const float*)d_in[10] };
    const float* bl[3] = { (const float*)d_in[5], (const float*)d_in[8], (const float*)d_in[11] };
    const float* Wr[3] = { (const float*)d_in[6], (const float*)d_in[9], (const float*)d_in[12] };

    // workspace carve-up (256B aligned)
    char* ws = (char*)d_ws;
    size_t off = 0;
    auto alloc = [&](size_t bytes) { size_t o = off; off = (off + bytes + 255) & ~(size_t)255; return o; };
    int*            cnt    = (int*)(ws + alloc((size_t)NL1 * 4));
    int*            offs   = (int*)(ws + alloc((size_t)(NL1 + 1) * 4));
    int*            cursor = (int*)(ws + alloc((size_t)NL1 * 4));
    int*            bsums  = (int*)(ws + alloc(1024 * 4));
    int*            srcs   = (int*)(ws + alloc((size_t)E0 * 4));
    unsigned short* Abf    = (unsigned short*)(ws + alloc((size_t)NL1 * 512 * 2));
    float*          h1     = (float*)(ws + alloc((size_t)NL1 * HID * 4));
    float*          h2     = (float*)(ws + alloc((size_t)NL2 * HID * 4));
    unsigned short* Wt0    = (unsigned short*)(ws + alloc((size_t)3 * 256 * 512 * 2));

    const int Ndst[3]    = { NL1, NL2, NL3 };
    const int Ec[3]      = { E0, E1, E2 };
    const float* hcur[3] = { x, h1, h2 };
    float* hnext[3]      = { h1, h2, (float*)d_out };

    // pack weights to bf16 n-major
    for (int l = 0; l < 3; ++l) {
        unsigned short* Wt = Wt0 + (size_t)l * 256 * 512;
        sage_prepw<<<(256 * 512 + 255) / 256, 256, 0, stream>>>(Wl[l], Wr[l], Wt);
    }

    for (int l = 0; l < 3; ++l) {
        const int Nd = Ndst[l], E = Ec[l];
        const int nb = (Nd + 255) / 256;          // <= 196 <= 1024

        // CSR build (int atomics + parallel scan)
        zero_i32<<<nb, 256, 0, stream>>>(cnt, Nd);
        sage_count<<<(int)cdivl(E, 256), 256, 0, stream>>>(ei[l], E, cnt);
        scan_block<<<nb, 256, 0, stream>>>(cnt, Nd, offs, bsums);
        scan_tops<<<1, 1024, 0, stream>>>(bsums, nb);
        scan_add<<<nb, 256, 0, stream>>>(offs, Nd, bsums, cursor);
        sage_reorder<<<(int)cdivl(E, 256), 256, 0, stream>>>(ei[l], E, cursor, srcs);

        // pull-mode mean aggregation + bf16 A build
        long at = (long)Nd * 64;
        sage_aggregate<<<(int)cdivl(at, 256), 256, 0, stream>>>(
            hcur[l], srcs, offs, Nd, E, Abf);

        // fused WMMA GEMM (+bias, +relu for layers 0/1)
        const unsigned short* Wt = Wt0 + (size_t)l * 256 * 512;
        int mtiles = (Nd + 15) / 16;
        int blocks = (mtiles * 4 + 7) / 8;
        sage_gemm<<<blocks, 256, 0, stream>>>(Abf, Wt, bl[l], hnext[l], Nd,
                                              (l != 2) ? 1 : 0);
    }
}
